// temporalAttention_6038724018672
// MI455X (gfx1250) — compile-verified, matrix operand order checked by reading, not run on hardware
//
#include <hip/hip_runtime.h>
#include <cstdint>

// Problem constants (from reference setup_inputs)
#define Bb 16
#define Tt 128
#define Nn 256
#define Ff 128

typedef __attribute__((ext_vector_type(16))) _Float16 v16h;
typedef __attribute__((ext_vector_type(8)))  _Float16 v8h;
typedef __attribute__((ext_vector_type(4)))  _Float16 v4h;
typedef __attribute__((ext_vector_type(8)))  float    v8f;

// Load one 16x32 f16 WMMA fragment from an LDS matrix with leading dim 128.
// Works for A (row = M tile base) and for B when B is stored transposed
// (rows = output column n, cols = contraction k). Per CDNA5 ISA:
// lane L holds row (L&15); element e holds K = (e<8 ? e : e+8) + 8*(L>>4).
__device__ __forceinline__ v16h ld_frag(const _Float16* m, int row, int k0) {
  const int lane = threadIdx.x & 31;
  const _Float16* p = m + (size_t)(row + (lane & 15)) * 128 + k0 + ((lane >> 4) << 3);
  v8h lo = *(const v8h*)p;         // K = 8g .. 8g+7
  v8h hi = *(const v8h*)(p + 16);  // K = 16+8g .. 16+8g+7
  return __builtin_shufflevector(lo, hi, 0,1,2,3,4,5,6,7,8,9,10,11,12,13,14,15);
}

__device__ __forceinline__ v8f wmma_acc(v16h a, v16h b, v8f c) {
  // (neg_a, A, neg_b, B, c_mod, C, reuse_a, reuse_b)
  return __builtin_amdgcn_wmma_f32_16x16x32_f16(false, a, false, b, (short)0, c, false, false);
}

// Async LDS<->global 16-byte movers (CDNA5, tracked by ASYNCcnt).
__device__ __forceinline__ void async_g2l_b128(void* lds, const void* g) {
  unsigned l = (unsigned)(uintptr_t)lds;                 // LDS addr = generic[31:0]
  unsigned long long a = (unsigned long long)(uintptr_t)g;
  asm volatile("global_load_async_to_lds_b128 %0, %1, off" :: "v"(l), "v"(a) : "memory");
}
__device__ __forceinline__ void async_l2g_b128(void* g, const void* lds) {
  unsigned l = (unsigned)(uintptr_t)lds;
  unsigned long long a = (unsigned long long)(uintptr_t)g;
  asm volatile("global_store_async_from_lds_b128 %0, %1, off" :: "v"(a), "v"(l) : "memory");
}
__device__ __forceinline__ void wait_async0() {
  asm volatile("s_wait_asynccnt 0x0" ::: "memory");
}

// ---------------------------------------------------------------------------
// Kernel 1: per-(b,n) head. x+te -> xs (f32 global, residual) + X (f16 LDS),
// QKV projections, causal attention, softmax, att@V -> ab [B,T,N,F] f16.
// Dynamic LDS = 224KB: [X 32K][W3/S 96K][Q/O 32K][K 32K][Vt 32K]
// ---------------------------------------------------------------------------
__global__ void __launch_bounds__(256) k_attn(
    const float* __restrict__ x, const float* __restrict__ te,
    const float* __restrict__ wq, const float* __restrict__ bq,
    const float* __restrict__ wk, const float* __restrict__ bk,
    const float* __restrict__ wv, const float* __restrict__ bv,
    float* __restrict__ xs, _Float16* __restrict__ ab)
{
  extern __shared__ char smem[];
  _Float16* Xl = (_Float16*)smem;                 // 32K (reused as Sb later)
  _Float16* Wl = (_Float16*)(smem + (32 << 10));  // 96K: wqT,wkT,wvT (reused as S f32)
  _Float16* Ql = (_Float16*)(smem + (128 << 10)); // 32K (reused as Ol later)
  _Float16* Kl = (_Float16*)(smem + (160 << 10)); // 32K
  _Float16* Vt = (_Float16*)(smem + (192 << 10)); // 32K (V transposed: [f][t])
  float*    S  = (float*)(smem + (32 << 10));     // 64K, overlaps Wl (weights dead)
  _Float16* Sb = Xl;                              // overlaps Xl (X dead)
  _Float16* Ol = Ql;                              // overlaps Ql (Q dead)

  const int tid = threadIdx.x;
  const int b = blockIdx.x >> 8;   // N = 256
  const int n = blockIdx.x & 255;

  // Warm L2 lines for the weights while we stream activations.
  if (tid < 128) {
    __builtin_prefetch(wq + tid * 128, 0, 0);
    __builtin_prefetch(wk + tid * 128, 0, 0);
    __builtin_prefetch(wv + tid * 128, 0, 0);
  }

  // ---- load X = x + te for this head, write residual xs, stage f16 X ----
  for (int i = tid; i < Tt * (Ff / 4); i += 256) {
    int t = i >> 5;            // Ff/4 == 32 chunks per row
    int c = (i & 31) << 2;
    size_t off = (((size_t)b * Tt + t) * Nn + n) * Ff + c;
    float4 xv = *(const float4*)(x + off);
    float4 tv = *(const float4*)(te + off);
    float4 s4;
    s4.x = xv.x + tv.x; s4.y = xv.y + tv.y; s4.z = xv.z + tv.z; s4.w = xv.w + tv.w;
    *(float4*)(xs + off) = s4;
    _Float16* dst = Xl + t * 128 + c;
    dst[0] = (_Float16)s4.x; dst[1] = (_Float16)s4.y;
    dst[2] = (_Float16)s4.z; dst[3] = (_Float16)s4.w;
  }
  // ---- stage wq/wk/wv transposed (Bt[n][k] = w[k][n]) as f16 ----
  for (int i = tid; i < 128 * 128; i += 256) {
    int f = i >> 7, g = i & 127;
    Wl[g * 128 + f]           = (_Float16)wq[i];
    Wl[16384 + g * 128 + f]   = (_Float16)wk[i];
    Wl[32768 + g * 128 + f]   = (_Float16)wv[i];
  }
  __syncthreads();

  const int wid   = tid >> 5;        // wave 0..7 owns tile-row tm = wid
  const int lane  = tid & 31;
  const int col   = lane & 15;       // within-tile N
  const int rbase = (lane >> 4) << 3;
  const int tm    = wid;

  // ---- Q/K/V projections: A frags hoisted; 2 output tiles per iteration ----
  {
    v16h xa[4];
    #pragma unroll
    for (int k = 0; k < 4; ++k) xa[k] = ld_frag(Xl, tm * 16, k << 5);
    for (int tn = 0; tn < 8; tn += 2) {
      v8f aq0 = {}, aq1 = {}, ak0 = {}, ak1 = {}, av0 = {}, av1 = {};
      #pragma unroll
      for (int k = 0; k < 4; ++k) {
        const int k0 = k << 5;
        v16h bq0 = ld_frag(Wl,         tn * 16,      k0);
        v16h bq1 = ld_frag(Wl,         tn * 16 + 16, k0);
        v16h bk0 = ld_frag(Wl + 16384, tn * 16,      k0);
        v16h bk1 = ld_frag(Wl + 16384, tn * 16 + 16, k0);
        v16h bv0 = ld_frag(Wl + 32768, tn * 16,      k0);
        v16h bv1 = ld_frag(Wl + 32768, tn * 16 + 16, k0);
        aq0 = wmma_acc(xa[k], bq0, aq0);
        aq1 = wmma_acc(xa[k], bq1, aq1);
        ak0 = wmma_acc(xa[k], bk0, ak0);
        ak1 = wmma_acc(xa[k], bk1, ak1);
        av0 = wmma_acc(xa[k], bv0, av0);
        av1 = wmma_acc(xa[k], bv1, av1);
      }
      float bqv0 = bq[tn * 16 + col],      bkv0 = bk[tn * 16 + col];
      float bvv0 = bv[tn * 16 + col];
      float bqv1 = bq[tn * 16 + 16 + col], bkv1 = bk[tn * 16 + 16 + col];
      float bvv1 = bv[tn * 16 + 16 + col];
      #pragma unroll
      for (int r = 0; r < 8; ++r) {
        int mrow  = tm * 16 + rbase + r;        // t index
        int nc0   = tn * 16 + col;              // f index (tile 0)
        int nc1   = tn * 16 + 16 + col;         // f index (tile 1)
        Ql[mrow * 128 + nc0] = (_Float16)(aq0[r] + bqv0);
        Ql[mrow * 128 + nc1] = (_Float16)(aq1[r] + bqv1);
        Kl[mrow * 128 + nc0] = (_Float16)(ak0[r] + bkv0);
        Kl[mrow * 128 + nc1] = (_Float16)(ak1[r] + bkv1);
        Vt[nc0 * 128 + mrow] = (_Float16)(av0[r] + bvv0);   // V stored transposed
        Vt[nc1 * 128 + mrow] = (_Float16)(av1[r] + bvv1);
      }
    }
  }
  __syncthreads();

  // ---- S = (Q @ K^T) / sqrt(16), causal mask to -32767 ----
  {
    v16h qa[4];
    #pragma unroll
    for (int k = 0; k < 4; ++k) qa[k] = ld_frag(Ql, tm * 16, k << 5);
    for (int tn = 0; tn < 8; tn += 2) {
      v8f ac0 = {}, ac1 = {};
      #pragma unroll
      for (int k = 0; k < 4; ++k) {
        const int k0 = k << 5;
        v16h b0 = ld_frag(Kl, tn * 16,      k0);
        v16h b1 = ld_frag(Kl, tn * 16 + 16, k0);
        ac0 = wmma_acc(qa[k], b0, ac0);
        ac1 = wmma_acc(qa[k], b1, ac1);
      }
      #pragma unroll
      for (int r = 0; r < 8; ++r) {
        int trow = tm * 16 + rbase + r;
        int s0 = tn * 16 + col, s1 = tn * 16 + 16 + col;
        float v0 = ac0[r] * 0.25f, v1 = ac1[r] * 0.25f;
        if (s0 > trow) v0 = -32767.0f;
        if (s1 > trow) v1 = -32767.0f;
        S[trow * 128 + s0] = v0;
        S[trow * 128 + s1] = v1;
      }
    }
  }
  __syncthreads();

  // ---- row softmax (f32 in LDS, float4-vectorized), f16 probs into Sb ----
  if (tid < 128) {
    float4* r4 = (float4*)(S + tid * 128);
    float mx = -3.4e38f;
    for (int s = 0; s < 32; ++s) {
      float4 v = r4[s];
      mx = fmaxf(mx, fmaxf(fmaxf(v.x, v.y), fmaxf(v.z, v.w)));
    }
    float sum = 0.0f;
    for (int s = 0; s < 32; ++s) {
      float4 v = r4[s];
      v.x = __expf(v.x - mx); v.y = __expf(v.y - mx);
      v.z = __expf(v.z - mx); v.w = __expf(v.w - mx);
      r4[s] = v;
      sum += v.x + v.y + v.z + v.w;
    }
    float inv = 1.0f / sum;
    v4h* o4 = (v4h*)(Sb + tid * 128);
    for (int s = 0; s < 32; ++s) {
      float4 v = r4[s];
      v4h h;
      h[0] = (_Float16)(v.x * inv); h[1] = (_Float16)(v.y * inv);
      h[2] = (_Float16)(v.z * inv); h[3] = (_Float16)(v.w * inv);
      o4[s] = h;
    }
  }
  __syncthreads();

  // ---- O = Sb @ V  -> Ol (LDS, reuse Ql) ----
  {
    v16h sa[4];
    #pragma unroll
    for (int k = 0; k < 4; ++k) sa[k] = ld_frag(Sb, tm * 16, k << 5);
    for (int tn = 0; tn < 8; tn += 2) {
      v8f ac0 = {}, ac1 = {};
      #pragma unroll
      for (int k = 0; k < 4; ++k) {
        const int k0 = k << 5;
        v16h b0 = ld_frag(Vt, tn * 16,      k0);
        v16h b1 = ld_frag(Vt, tn * 16 + 16, k0);
        ac0 = wmma_acc(sa[k], b0, ac0);
        ac1 = wmma_acc(sa[k], b1, ac1);
      }
      #pragma unroll
      for (int r = 0; r < 8; ++r) {
        int trow = tm * 16 + rbase + r;
        Ol[trow * 128 + tn * 16 + col]      = (_Float16)ac0[r];
        Ol[trow * 128 + tn * 16 + 16 + col] = (_Float16)ac1[r];
      }
    }
  }
  __syncthreads();

  // ---- coalesced async LDS->global store of ab rows [B,T,N,F] ----
  for (int i = tid; i < 128 * 16; i += 256) {
    int t = i >> 4; int c = (i & 15) << 3;
    size_t off = (((size_t)b * Tt + t) * Nn + n) * Ff + c;
    async_l2g_b128(ab + off, Ol + t * 128 + c);
  }
  wait_async0();
}

// ---------------------------------------------------------------------------
// Kernel 2: o = ab @ wo + bo + xs; y = LayerNorm(o). 128 tokens per block.
// Dynamic LDS = 128KB: [A 32K][WoT 32K][O f32 64K]
// ---------------------------------------------------------------------------
__global__ void __launch_bounds__(256) k_proj_ln(
    const _Float16* __restrict__ ab, const float* __restrict__ wo,
    const float* __restrict__ bo, const float* __restrict__ xs,
    float* __restrict__ y, _Float16* __restrict__ yb)
{
  extern __shared__ char smem[];
  _Float16* Al = (_Float16*)smem;
  _Float16* Wt = (_Float16*)(smem + (32 << 10));
  float*    O  = (float*)(smem + (64 << 10));
  const int tid = threadIdx.x;
  const size_t row0 = (size_t)blockIdx.x * 128;

  if (tid < 128) __builtin_prefetch(wo + tid * 128, 0, 0);

  // Async global->LDS copy of the f16 activation tile.
  for (int i = tid; i < 128 * 16; i += 256) {
    int t = i >> 4; int c = (i & 15) << 3;
    async_g2l_b128(Al + t * 128 + c, ab + (row0 + t) * Ff + c);
  }
  for (int i = tid; i < 128 * 128; i += 256) {
    int f = i >> 7, g = i & 127;
    Wt[g * 128 + f] = (_Float16)wo[i];
  }
  wait_async0();
  __syncthreads();

  const int wid = tid >> 5, lane = tid & 31;
  const int col = lane & 15, rbase = (lane >> 4) << 3;
  {
    v16h a[4];
    #pragma unroll
    for (int k = 0; k < 4; ++k) a[k] = ld_frag(Al, wid * 16, k << 5);
    for (int tn = 0; tn < 8; tn += 2) {
      v8f ac0 = {}, ac1 = {};
      #pragma unroll
      for (int k = 0; k < 4; ++k) {
        const int k0 = k << 5;
        v16h b0 = ld_frag(Wt, tn * 16,      k0);
        v16h b1 = ld_frag(Wt, tn * 16 + 16, k0);
        ac0 = wmma_acc(a[k], b0, ac0);
        ac1 = wmma_acc(a[k], b1, ac1);
      }
      float bo0 = bo[tn * 16 + col], bo1 = bo[tn * 16 + 16 + col];
      #pragma unroll
      for (int r = 0; r < 8; ++r) {
        int m = wid * 16 + rbase + r;
        int n0 = tn * 16 + col, n1 = tn * 16 + 16 + col;
        O[m * 128 + n0] = ac0[r] + bo0 + xs[(row0 + m) * Ff + n0];
        O[m * 128 + n1] = ac1[r] + bo1 + xs[(row0 + m) * Ff + n1];
      }
    }
  }
  __syncthreads();

  if (tid < 128) {
    float4* r4 = (float4*)(O + tid * 128);
    float mu = 0.0f;
    for (int c = 0; c < 32; ++c) { float4 v = r4[c]; mu += v.x + v.y + v.z + v.w; }
    mu *= (1.0f / 128.0f);
    float var = 0.0f;
    for (int c = 0; c < 32; ++c) {
      float4 v = r4[c];
      float dx = v.x - mu, dy = v.y - mu, dz = v.z - mu, dw = v.w - mu;
      var += dx * dx + dy * dy + dz * dz + dw * dw;
    }
    var *= (1.0f / 128.0f);
    float rstd = rsqrtf(var + 1e-5f);
    size_t ro = (row0 + tid) * Ff;
    for (int c = 0; c < 32; ++c) {
      float4 v = r4[c];
      float4 o4;
      o4.x = (v.x - mu) * rstd; o4.y = (v.y - mu) * rstd;
      o4.z = (v.z - mu) * rstd; o4.w = (v.w - mu) * rstd;
      *(float4*)(y + ro + c * 4) = o4;
      v4h h; h[0] = (_Float16)o4.x; h[1] = (_Float16)o4.y;
      h[2] = (_Float16)o4.z; h[3] = (_Float16)o4.w;
      *(v4h*)(yb + ro + c * 4) = h;
    }
  }
}

// ---------------------------------------------------------------------------
// Kernel 3: h = relu(yb@w1+b1); out = LayerNorm(h@w2+b2 + y). Fully fused FFN.
// Dynamic LDS = 192KB: [A 32K][W1T 32K][W2T 32K][H 32K][O f32 64K]
// ---------------------------------------------------------------------------
__global__ void __launch_bounds__(256) k_ffn_ln(
    const _Float16* __restrict__ yb, const float* __restrict__ w1,
    const float* __restrict__ b1, const float* __restrict__ w2,
    const float* __restrict__ b2, const float* __restrict__ y,
    float* __restrict__ out)
{
  extern __shared__ char smem[];
  _Float16* Al  = (_Float16*)smem;
  _Float16* W1t = (_Float16*)(smem + (32 << 10));
  _Float16* W2t = (_Float16*)(smem + (64 << 10));
  _Float16* Hl  = (_Float16*)(smem + (96 << 10));
  float*    O   = (float*)(smem + (128 << 10));
  const int tid = threadIdx.x;
  const size_t row0 = (size_t)blockIdx.x * 128;

  if (tid < 128) {
    __builtin_prefetch(w1 + tid * 128, 0, 0);
    __builtin_prefetch(w2 + tid * 128, 0, 0);
  }

  for (int i = tid; i < 128 * 16; i += 256) {
    int t = i >> 4; int c = (i & 15) << 3;
    async_g2l_b128(Al + t * 128 + c, yb + (row0 + t) * Ff + c);
  }
  for (int i = tid; i < 128 * 128; i += 256) {
    int f = i >> 7, g = i & 127;
    W1t[g * 128 + f] = (_Float16)w1[i];
    W2t[g * 128 + f] = (_Float16)w2[i];
  }
  wait_async0();
  __syncthreads();

  const int wid = tid >> 5, lane = tid & 31;
  const int col = lane & 15, rbase = (lane >> 4) << 3;

  // H = relu(A @ w1 + b1), kept in LDS (never touches HBM)
  {
    v16h a[4];
    #pragma unroll
    for (int k = 0; k < 4; ++k) a[k] = ld_frag(Al, wid * 16, k << 5);
    for (int tn = 0; tn < 8; tn += 2) {
      v8f ac0 = {}, ac1 = {};
      #pragma unroll
      for (int k = 0; k < 4; ++k) {
        const int k0 = k << 5;
        v16h b0 = ld_frag(W1t, tn * 16,      k0);
        v16h b1 = ld_frag(W1t, tn * 16 + 16, k0);
        ac0 = wmma_acc(a[k], b0, ac0);
        ac1 = wmma_acc(a[k], b1, ac1);
      }
      float bb0 = b1[tn * 16 + col], bb1 = b1[tn * 16 + 16 + col];
      #pragma unroll
      for (int r = 0; r < 8; ++r) {
        int m = wid * 16 + rbase + r;
        float h0 = ac0[r] + bb0, h1 = ac1[r] + bb1;
        Hl[m * 128 + tn * 16 + col]      = (_Float16)(h0 > 0.0f ? h0 : 0.0f);
        Hl[m * 128 + tn * 16 + 16 + col] = (_Float16)(h1 > 0.0f ? h1 : 0.0f);
      }
    }
  }
  __syncthreads();

  // O = H @ w2 + b2 + y
  {
    v16h a[4];
    #pragma unroll
    for (int k = 0; k < 4; ++k) a[k] = ld_frag(Hl, wid * 16, k << 5);
    for (int tn = 0; tn < 8; tn += 2) {
      v8f ac0 = {}, ac1 = {};
      #pragma unroll
      for (int k = 0; k < 4; ++k) {
        const int k0 = k << 5;
        v16h b0 = ld_frag(W2t, tn * 16,      k0);
        v16h b1 = ld_frag(W2t, tn * 16 + 16, k0);
        ac0 = wmma_acc(a[k], b0, ac0);
        ac1 = wmma_acc(a[k], b1, ac1);
      }
      float bb0 = b2[tn * 16 + col], bb1 = b2[tn * 16 + 16 + col];
      #pragma unroll
      for (int r = 0; r < 8; ++r) {
        int m = wid * 16 + rbase + r;
        int n0 = tn * 16 + col, n1 = tn * 16 + 16 + col;
        O[m * 128 + n0] = ac0[r] + bb0 + y[(row0 + m) * Ff + n0];
        O[m * 128 + n1] = ac1[r] + bb1 + y[(row0 + m) * Ff + n1];
      }
    }
  }
  __syncthreads();

  if (tid < 128) {
    float4* r4 = (float4*)(O + tid * 128);
    float mu = 0.0f;
    for (int c = 0; c < 32; ++c) { float4 v = r4[c]; mu += v.x + v.y + v.z + v.w; }
    mu *= (1.0f / 128.0f);
    float var = 0.0f;
    for (int c = 0; c < 32; ++c) {
      float4 v = r4[c];
      float dx = v.x - mu, dy = v.y - mu, dz = v.z - mu, dw = v.w - mu;
      var += dx * dx + dy * dy + dz * dz + dw * dw;
    }
    var *= (1.0f / 128.0f);
    float rstd = rsqrtf(var + 1e-5f);
    size_t ro = (row0 + tid) * Ff;
    for (int c = 0; c < 32; ++c) {
      float4 v = r4[c];
      float4 o4;
      o4.x = (v.x - mu) * rstd; o4.y = (v.y - mu) * rstd;
      o4.z = (v.z - mu) * rstd; o4.w = (v.w - mu) * rstd;
      *(float4*)(out + ro + c * 4) = o4;
    }
  }
}

// ---------------------------------------------------------------------------
extern "C" void kernel_launch(void* const* d_in, const int* in_sizes, int n_in,
                              void* d_out, int out_size, void* d_ws, size_t ws_size,
                              hipStream_t stream) {
  (void)in_sizes; (void)n_in; (void)out_size; (void)ws_size;
  const float* x  = (const float*)d_in[0];
  const float* te = (const float*)d_in[1];
  const float* wq = (const float*)d_in[2];
  const float* bq = (const float*)d_in[3];
  const float* wk = (const float*)d_in[4];
  const float* bk = (const float*)d_in[5];
  const float* wv = (const float*)d_in[6];
  const float* bv = (const float*)d_in[7];
  const float* wo = (const float*)d_in[8];
  const float* bo = (const float*)d_in[9];
  const float* w1 = (const float*)d_in[10];
  const float* b1 = (const float*)d_in[11];
  const float* w2 = (const float*)d_in[12];
  const float* b2 = (const float*)d_in[13];

  // Workspace carve (TOK*F = 16*128*256*128 = 67,108,864 elements):
  //   xs  f32  268MB (x+te residual) | y f32 268MB | ab f16 134MB | yb f16 134MB
  char* ws = (char*)d_ws;
  const size_t TOKF = (size_t)Bb * Tt * Nn * Ff;
  float*    xs  = (float*)ws;
  float*    yy  = (float*)(ws + TOKF * 4);
  _Float16* ab  = (_Float16*)(ws + TOKF * 8);
  _Float16* ybb = (_Float16*)(ws + TOKF * 8 + TOKF * 2);

  // Opt in to >64KB dynamic LDS (CDNA5 WGP supports 320KB).
  (void)hipFuncSetAttribute((const void*)k_attn,
        hipFuncAttributeMaxDynamicSharedMemorySize, 224 << 10);
  (void)hipFuncSetAttribute((const void*)k_proj_ln,
        hipFuncAttributeMaxDynamicSharedMemorySize, 128 << 10);
  (void)hipFuncSetAttribute((const void*)k_ffn_ln,
        hipFuncAttributeMaxDynamicSharedMemorySize, 192 << 10);

  k_attn<<<Bb * Nn, 256, 224 << 10, stream>>>(x, te, wq, bq, wk, bk, wv, bv, xs, ab);
  k_proj_ln<<<(Bb * Tt * Nn) / 128, 256, 128 << 10, stream>>>(ab, wo, bo, xs, yy, ybb);
  k_ffn_ln<<<(Bb * Tt * Nn) / 128, 256, 192 << 10, stream>>>(ybb, w1, b1, w2, b2, yy,
                                                             (float*)d_out);
}
